// MultiHeadAttentionLayer_52939766890939
// MI455X (gfx1250) — matrix operand order repaired and checked by reading, not compiled, not run on hardware
//
#include <hip/hip_runtime.h>
#include <hip/hip_bf16.h>

#define N_NODES 100000
#define N_EDGES 1600000
#define IN_DIM 128
#define EDGE_DIM 64
#define HD 64          // H*D
#define OUT_DIM 8

typedef __attribute__((ext_vector_type(16))) __bf16 v16bf;
typedef __attribute__((ext_vector_type(8)))  float  v8f;

__device__ __forceinline__ __bf16 f2bf(float f) { return (__bf16)f; }

// ---- WMMA bf16 fragment helpers (wave32, 16x16x32) -----------------------
// A-fragment: 16x32 tile (rows m, cols k) from row-major f32, leading dim ld.
// Lane l: row m = l&15, K quarter kq = (l>>4)*8; elems k in [kq,kq+8) and [kq+16,kq+24).
__device__ __forceinline__ v16bf load_a_frag(const float* __restrict__ base, int ld, int lane) {
  const int m  = lane & 15;
  const int kq = (lane >> 4) * 8;
  const float* p = base + (size_t)m * ld + kq;
  v16bf a;
#pragma unroll
  for (int i = 0; i < 8; ++i) {
    a[i]     = f2bf(p[i]);
    a[8 + i] = f2bf(p[16 + i]);
  }
  return a;
}

// Pre-packed B fragment: 512 bf16 per fragment, lane-major (lane*16 contiguous).
__device__ __forceinline__ v16bf load_b_packed(const __bf16* __restrict__ packed,
                                               int frag, int lane) {
  return *reinterpret_cast<const v16bf*>(packed + (size_t)frag * 512 + lane * 16);
}

// ---- Weight packing: W[K x N] f32 -> per-fragment bf16 in B-frag layout ---
// frag f = kc*(N/16) + nt ; element (f, lane l, i) = W[kc*32 + (l>>4)*8 + (i&7) + (i>>3)*16][nt*16 + (l&15)]
__global__ void pack_weight_kernel(const float* __restrict__ W, __bf16* __restrict__ dst,
                                   int N, int total) {
  int idx = blockIdx.x * blockDim.x + threadIdx.x;
  if (idx >= total) return;
  const int i  = idx & 15;
  const int l  = (idx >> 4) & 31;
  const int f  = idx >> 9;
  const int ntiles = N >> 4;
  const int nt = f % ntiles;
  const int kc = f / ntiles;
  const int k  = kc * 32 + ((l >> 4) << 3) + (i & 7) + ((i >> 3) << 4);
  const int n  = nt * 16 + (l & 15);
  dst[idx] = f2bf(W[(size_t)k * N + n]);
}

// Wo is 64x8 -> pad N to 16 with zeros (2 fragments, kc = 0,1)
__global__ void pack_wo_kernel(const float* __restrict__ Wo, __bf16* __restrict__ dst) {
  int idx = blockIdx.x * blockDim.x + threadIdx.x;
  if (idx >= 1024) return;
  const int i  = idx & 15;
  const int l  = (idx >> 4) & 31;
  const int kc = idx >> 9;
  const int k  = kc * 32 + ((l >> 4) << 3) + (i & 7) + ((i >> 3) << 4);
  const int n  = l & 15;
  dst[idx] = (n < OUT_DIM) ? f2bf(Wo[(size_t)k * OUT_DIM + n]) : f2bf(0.0f);
}

// ---- Kernel 1: Q/K/V node projections (48 WMMAs / 16-node tile) ----------
__global__ __launch_bounds__(128) void qkv_proj_kernel(
    const float* __restrict__ x,
    const __bf16* __restrict__ pWq, const __bf16* __restrict__ pWk,
    const __bf16* __restrict__ pWv,
    const float* __restrict__ bq, const float* __restrict__ bk,
    const float* __restrict__ bv,
    float* __restrict__ Q, float* __restrict__ K, float* __restrict__ V)
{
  const int wave = blockIdx.x * (blockDim.x >> 5) + (threadIdx.x >> 5);
  if (wave >= N_NODES / 16) return;          // wave-uniform guard
  const int lane = threadIdx.x & 31;
  const int row0 = wave * 16;

  const __bf16* PW[3] = {pWq, pWk, pWv};
  v8f acc[3][4] = {};

#pragma unroll
  for (int kc = 0; kc < 4; ++kc) {           // K = 128 in chunks of 32
    v16bf a = load_a_frag(x + (size_t)row0 * IN_DIM + kc * 32, IN_DIM, lane);
#pragma unroll
    for (int w = 0; w < 3; ++w) {
#pragma unroll
      for (int nt = 0; nt < 4; ++nt) {       // N = 64 in tiles of 16
        v16bf b = load_b_packed(PW[w], kc * 4 + nt, lane);
        acc[w][nt] = __builtin_amdgcn_wmma_f32_16x16x32_bf16(
            false, a, false, b, (short)0, acc[w][nt], false, false);
      }
    }
  }

  const int n  = lane & 15;
  const int mb = (lane >> 4) * 8;
  float*       O[3] = {Q, K, V};
  const float* B[3] = {bq, bk, bv};
#pragma unroll
  for (int w = 0; w < 3; ++w) {
#pragma unroll
    for (int nt = 0; nt < 4; ++nt) {
      const float bias = B[w][nt * 16 + n];
#pragma unroll
      for (int v = 0; v < 8; ++v)
        O[w][(size_t)(row0 + mb + v) * HD + nt * 16 + n] = acc[w][nt][v] + bias;
    }
  }
}

// ---- Kernel 2: fused edge pipeline (grid-stride, weights in registers) ---
__global__ __launch_bounds__(128) void edge_fused_kernel(
    const float* __restrict__ edge_attr,
    const __bf16* __restrict__ pWe, const float* __restrict__ be,
    const __bf16* __restrict__ pWo, const float* __restrict__ bo,
    const int* __restrict__ src, const int* __restrict__ dst,
    const float* __restrict__ Q, const float* __restrict__ K,
    const float* __restrict__ V,
    float* __restrict__ out, float* __restrict__ counts)
{
  __shared__ float sbuf[4][16][8];           // per-wave [edge-in-tile][head]
  const int wid  = threadIdx.x >> 5;
  const int lane = threadIdx.x & 31;
  const int nwaves = gridDim.x * (blockDim.x >> 5);
  const int wave0  = blockIdx.x * (blockDim.x >> 5) + wid;

  // hoist loop-invariant weight fragments + biases into registers
  v16bf bWe[8], bWo[2];
#pragma unroll
  for (int f = 0; f < 8; ++f) bWe[f] = load_b_packed(pWe, f, lane);
#pragma unroll
  for (int f = 0; f < 2; ++f) bWo[f] = load_b_packed(pWo, f, lane);

  const int m = lane & 15;                   // edge within tile
  const int g = lane >> 4;                   // head group: heads g*4 .. g*4+3
  float beL[4];
#pragma unroll
  for (int nt = 0; nt < 4; ++nt) beL[nt] = be[nt * 16 + m];  // n==m (lane&15)
  const float boL = bo[m & 7];

  for (int tile = wave0; tile < N_EDGES / 16; tile += nwaves) {
    const int e0 = tile * 16;

    // prefetch next tile's edge_attr (4 KB = 32 lanes x 128 B)
    {
      const int nt_tile = tile + nwaves;
      if (nt_tile < N_EDGES / 16) {
        const char* pf = (const char*)(edge_attr + (size_t)nt_tile * 16 * EDGE_DIM)
                         + lane * 128;
        __builtin_prefetch(pf, 0, 1);
      }
    }

    // --- E projection + per-head squared norms ---
    v8f eacc[4] = {};
#pragma unroll
    for (int kc = 0; kc < 2; ++kc) {         // K = 64 in chunks of 32
      v16bf a = load_a_frag(edge_attr + (size_t)e0 * EDGE_DIM + kc * 32,
                            EDGE_DIM, lane);
#pragma unroll
      for (int nt = 0; nt < 4; ++nt) {
        eacc[nt] = __builtin_amdgcn_wmma_f32_16x16x32_bf16(
            false, a, false, bWe[kc * 4 + nt], (short)0, eacc[nt], false, false);
      }
    }
    {
      const int mb   = (lane >> 4) * 8;
      const int hsub = (lane >> 3) & 1;      // head within 16-wide n-tile
#pragma unroll
      for (int nt = 0; nt < 4; ++nt) {
#pragma unroll
        for (int v = 0; v < 8; ++v) {
          float ev = eacc[nt][v] + beL[nt];
          float s2 = ev * ev;
          s2 += __shfl_xor(s2, 1, 32);
          s2 += __shfl_xor(s2, 2, 32);
          s2 += __shfl_xor(s2, 4, 32);       // sum over the 8 lanes of one head
          if ((lane & 7) == 0)
            sbuf[wid][mb + v][nt * 2 + hsub] = s2;
        }
      }
    }

    // --- per-edge scores + softmax over 8 heads ---
    const int s_node = src[e0 + m];
    const int d_node = dst[e0 + m];
    {
      const float* qrow = Q + (size_t)s_node * HD + g * 32;
      const float* krow = K + (size_t)d_node * HD + g * 32;
      float sc[4];
#pragma unroll
      for (int hh = 0; hh < 4; ++hh) {
        float acc = 0.f;
#pragma unroll
        for (int d = 0; d < 8; ++d) acc += qrow[hh * 8 + d] * krow[hh * 8 + d];
        sc[hh] = (acc + sbuf[wid][m][g * 4 + hh]) * 0.35355339059f;  // 1/sqrt(D)
      }
      float mx = fmaxf(fmaxf(sc[0], sc[1]), fmaxf(sc[2], sc[3]));
      mx = fmaxf(mx, __shfl_xor(mx, 16, 32));
      float sum = 0.f;
#pragma unroll
      for (int hh = 0; hh < 4; ++hh) { sc[hh] = __expf(sc[hh] - mx); sum += sc[hh]; }
      sum += __shfl_xor(sum, 16, 32);
      const float inv = 1.0f / sum;
#pragma unroll
      for (int hh = 0; hh < 4; ++hh) sbuf[wid][m][g * 4 + hh] = sc[hh] * inv;
    }

    // --- (scores * V[dst]) @ Wo + bo, scatter-add over src ---
    v8f oacc = {};
    {
      const int kq = g * 8;                  // A-frag K quarter for row m
      const float* vrow = V + (size_t)d_node * HD;
#pragma unroll
      for (int kc = 0; kc < 2; ++kc) {
        const int kbase = kc * 32 + kq;
        v16bf a;
#pragma unroll
        for (int i = 0; i < 8; ++i) {
          const int k1 = kbase + i;
          const int k2 = kbase + 16 + i;
          a[i]     = f2bf(vrow[k1] * sbuf[wid][m][k1 >> 3]);
          a[8 + i] = f2bf(vrow[k2] * sbuf[wid][m][k2 >> 3]);
        }
        oacc = __builtin_amdgcn_wmma_f32_16x16x32_bf16(
            false, a, false, bWo[kc], (short)0, oacc, false, false);
      }
    }
    {
      const int nB = lane & 15;
      const int mb = (lane >> 4) * 8;
      if (nB < OUT_DIM) {
#pragma unroll
        for (int v = 0; v < 8; ++v) {
          const int sidx = src[e0 + mb + v];
          atomicAdd(&out[(size_t)sidx * OUT_DIM + nB], oacc[v] + boL);
        }
      }
      if (lane < 16) atomicAdd(&counts[s_node], 1.0f);
    }
  }
}

// ---- Kernel 3: scatter-mean finalize -------------------------------------
__global__ void finalize_kernel(float* __restrict__ out,
                                const float* __restrict__ counts, int total)
{
  int i = blockIdx.x * blockDim.x + threadIdx.x;
  if (i < total) out[i] = out[i] / fmaxf(counts[i >> 3], 1.0f);
}

extern "C" void kernel_launch(void* const* d_in, const int* in_sizes, int n_in,
                              void* d_out, int out_size, void* d_ws, size_t ws_size,
                              hipStream_t stream) {
  const float* x  = (const float*)d_in[0];
  const float* ea = (const float*)d_in[1];
  const float* Wq = (const float*)d_in[2];
  const float* bq = (const float*)d_in[3];
  const float* Wk = (const float*)d_in[4];
  const float* bk = (const float*)d_in[5];
  const float* Wv = (const float*)d_in[6];
  const float* bv = (const float*)d_in[7];
  const float* We = (const float*)d_in[8];
  const float* be = (const float*)d_in[9];
  const float* Wo = (const float*)d_in[10];
  const float* bo = (const float*)d_in[11];
  const int*   ei = (const int*)d_in[12];
  float* out = (float*)d_out;

  // workspace layout: packed bf16 weights (64 KB slot) | Q | K | V | counts
  __bf16* pWq = (__bf16*)d_ws;                 // 16 frags * 512
  __bf16* pWk = pWq + 16 * 512;
  __bf16* pWv = pWk + 16 * 512;
  __bf16* pWe = pWv + 16 * 512;                // 8 frags
  __bf16* pWo = pWe + 8 * 512;                 // 2 frags
  float* Q      = (float*)((char*)d_ws + 65536);
  float* K      = Q + (size_t)N_NODES * HD;
  float* V      = K + (size_t)N_NODES * HD;
  float* counts = V + (size_t)N_NODES * HD;

  hipMemsetAsync(out,    0, (size_t)out_size * sizeof(float), stream);
  hipMemsetAsync(counts, 0, (size_t)N_NODES  * sizeof(float), stream);

  // pack weights to bf16 fragment layout
  pack_weight_kernel<<<(16 * 512 + 255) / 256, 256, 0, stream>>>(Wq, pWq, HD, 16 * 512);
  pack_weight_kernel<<<(16 * 512 + 255) / 256, 256, 0, stream>>>(Wk, pWk, HD, 16 * 512);
  pack_weight_kernel<<<(16 * 512 + 255) / 256, 256, 0, stream>>>(Wv, pWv, HD, 16 * 512);
  pack_weight_kernel<<<(8  * 512 + 255) / 256, 256, 0, stream>>>(We, pWe, HD, 8 * 512);
  pack_wo_kernel<<<4, 256, 0, stream>>>(Wo, pWo);

  {
    const int waves  = N_NODES / 16;               // 6250 tiles
    const int blocks = (waves + 3) / 4;            // 4 waves / block
    qkv_proj_kernel<<<blocks, 128, 0, stream>>>(x, pWq, pWk, pWv, bq, bk, bv, Q, K, V);
  }
  {
    const int blocks = 1250;                       // 5000 waves, ~20 tiles each
    edge_fused_kernel<<<blocks, 128, 0, stream>>>(ea, pWe, be, pWo, bo,
        ei, ei + N_EDGES, Q, K, V, out, counts);
  }
  {
    const int total = N_NODES * OUT_DIM;
    finalize_kernel<<<(total + 255) / 256, 256, 0, stream>>>(out, counts, total);
  }
}
